// ResidualBlock_8203387535370
// MI455X (gfx1250) — compile-verified, matrix operand order checked by reading, not run on hardware
//
#include <hip/hip_runtime.h>
#include <hip/hip_bf16.h>

typedef __bf16 bf16;
typedef __attribute__((ext_vector_type(16))) __bf16 v16bf;
typedef __attribute__((ext_vector_type(8)))  __bf16 v8bf;
typedef __attribute__((ext_vector_type(8)))  float  v8f;

// ---------------------------------------------------------------------------
// WMMA helpers (CDNA5 bf16 16x16x32, f32 accumulate)
// ---------------------------------------------------------------------------
__device__ __forceinline__ v8f wmma_bf16(v16bf a, v16bf b, v8f c) {
  return __builtin_amdgcn_wmma_f32_16x16x32_bf16(
      /*neg_a=*/false, a, /*neg_b=*/false, b,
      /*c_mod=*/(short)0, c, /*reuse_a=*/false, /*reuse_b=*/false);
}

// Fragment element (r, k) = s[r*ld + k]  (r = A-row / B-column = lane%16).
// K layout per lane-half: VGPR j holds K = (j/4)*16 + half*8 + (j%4)*2 (+1),
// i.e. two contiguous 8-element runs at r*ld + half*8 and r*ld + 16 + half*8.
// Requires: base and r*ld 16-byte aligned (all our ld's are multiples of 8).
__device__ __forceinline__ v16bf frag_rk(const bf16* s, int ld) {
  const int lane = threadIdx.x & 31;
  const int r = lane & 15, hh = lane >> 4;
  const v8bf lo = *(const v8bf*)(s + r * ld + hh * 8);
  const v8bf hi = *(const v8bf*)(s + r * ld + 16 + hh * 8);
  v16bf f;
#pragma unroll
  for (int e = 0; e < 8; ++e) { f[e] = lo[e]; f[8 + e] = hi[e]; }
  return f;
}

// wave32: reduce across the 16-lane group (xor<16 stays within the half)
__device__ __forceinline__ float grp16_max(float v) {
  v = fmaxf(v, __shfl_xor(v, 1, 32));
  v = fmaxf(v, __shfl_xor(v, 2, 32));
  v = fmaxf(v, __shfl_xor(v, 4, 32));
  v = fmaxf(v, __shfl_xor(v, 8, 32));
  return v;
}
__device__ __forceinline__ float grp16_sum(float v) {
  v += __shfl_xor(v, 1, 32);
  v += __shfl_xor(v, 2, 32);
  v += __shfl_xor(v, 4, 32);
  v += __shfl_xor(v, 8, 32);
  return v;
}

// ---------------------------------------------------------------------------
// Kernel A: 3x3 conv (64->64) + PReLU, implicit GEMM via WMMA.
// Grid: 16*128 blocks (one per (n, output row h)); 256 threads = 8 waves.
// Input rows staged as [3][130][72] (w-major rows, ci contiguous) so the
// B-operand is fragment-contiguous; weights as [64][584], k = tap*64+ci.
// ---------------------------------------------------------------------------
#define CONV_LDW 584   // 576 + 8 pad
#define CONV_LDX 72    // 64 + 8 pad

__global__ __launch_bounds__(256)
void conv_prelu_kernel(const float* __restrict__ x, const float* __restrict__ wgt,
                       const float* __restrict__ bias, const float* __restrict__ alpha,
                       float* __restrict__ r) {
  extern __shared__ char smem[];
  bf16* Xl = (bf16*)smem;                 // [3*130][72]
  bf16* Wl = Xl + 3 * 130 * CONV_LDX;     // [64][584]

  const int nb = blockIdx.x;
  const int n = nb >> 7, h = nb & 127;
  const int tid = threadIdx.x, lane = tid & 31, wave = tid >> 5;

  // stage input rows (w fastest in the loop index for coalesced global reads)
  for (int i = tid; i < 64 * 3 * 130; i += 256) {
    int ci = i / 390, rem = i - ci * 390;
    int rr = rem / 130, wp = rem - rr * 130;
    int hs = h + rr - 1, ws = wp - 1;
    float v = 0.0f;
    if (hs >= 0 && hs < 128 && ws >= 0 && ws < 128)
      v = x[((n * 64 + ci) * 128 + hs) * 128 + ws];
    Xl[(rr * 130 + wp) * CONV_LDX + ci] = (bf16)v;
  }
  for (int i = tid; i < 64 * 576; i += 256) {
    int co = i / 576, k = i - co * 576;
    int tap = k >> 6, ci = k & 63;
    Wl[co * CONV_LDW + k] = (bf16)wgt[(co * 64 + ci) * 9 + tap];
  }
  __syncthreads();

  const int w0 = wave * 16;  // pixel tile of this wave
  v8f acc[4] = {};
  for (int c = 0; c < 18; ++c) {      // K chunks of 32: tap-major, 2 per tap
    int tap = c >> 1, ci0 = (c & 1) << 5;
    int kh = tap / 3, kw = tap - kh * 3;
    v16bf b = frag_rk(Xl + (kh * 130 + w0 + kw) * CONV_LDX + ci0, CONV_LDX);
#pragma unroll
    for (int mt = 0; mt < 4; ++mt) {
      v16bf a = frag_rk(Wl + (mt * 16) * CONV_LDW + tap * 64 + ci0, CONV_LDW);
      acc[mt] = wmma_bf16(a, b, acc[mt]);
    }
  }

  const float al = alpha[0];
  const int pix = w0 + (lane & 15), hh = lane >> 4;
#pragma unroll
  for (int mt = 0; mt < 4; ++mt) {
#pragma unroll
    for (int i = 0; i < 8; ++i) {
      int co = mt * 16 + i + 8 * hh;
      float v = acc[mt][i] + bias[co];
      v = (v >= 0.0f) ? v : al * v;
      r[((n * 64 + co) * 128 + h) * 128 + pix] = v;
    }
  }
}

// ---------------------------------------------------------------------------
// Kernel B: long-range SA, in-place on r.
// Grid: 1024 blocks (one per group g=(n,ph,pw)); 256 threads = 8 waves.
// Group = [C=64][n=256 positions]; KC=32 => one bf16 K-chunk per S tile.
// LDS orientations chosen so every WMMA operand is fragment-contiguous:
//   XbT[pos][c], Qt[pos][kc], Kt[pos][kc], Vb[v][pos], Ab[np][m], Ct[pos][v].
// ---------------------------------------------------------------------------
#define LDW  72    // weight row stride (K=64  + 8 pad)
#define LDP  72    // XbT/Ct row stride (64 channels + 8 pad)
#define LDQK 40    // Qt/Kt row stride  (KC=32 + 8 pad)
#define LDN  264   // Vb/Ab row stride  (256 positions + 8 pad)

__global__ __launch_bounds__(256)
void isa_long_kernel(float* __restrict__ r,
                     const float* __restrict__ qwm, const float* __restrict__ qb,
                     const float* __restrict__ kwm, const float* __restrict__ kb,
                     const float* __restrict__ vwm, const float* __restrict__ vb,
                     const float* __restrict__ owm, const float* __restrict__ ob) {
  extern __shared__ char smem[];
  bf16* Wq  = (bf16*)smem;           // 32 x LDW
  bf16* Wk  = Wq + 32 * LDW;
  bf16* Wv  = Wk + 32 * LDW;         // 64 x LDW
  bf16* Wo  = Wv + 64 * LDW;
  bf16* XbT = Wo + 64 * LDW;         // 256 x LDP   [pos][c]  (aliased by Ct)
  bf16* Qt  = XbT + 256 * LDP;       // 256 x LDQK  [pos][kc]
  bf16* Kt  = Qt + 256 * LDQK;       // 256 x LDQK
  bf16* Vb  = Kt + 256 * LDQK;       // 64 x LDN    [v][pos]
  bf16* Ab  = Vb + 64 * LDN;         // 256 x LDN   [np][m]  unnormalized attn
  bf16* Ct  = XbT;                   // 256 x LDP   [pos][v]

  const int g = blockIdx.x;
  const int n = g >> 6, ph = (g >> 3) & 7, pw = g & 7;
  const int tid = threadIdx.x, lane = tid & 31, wave = tid >> 5;

  for (int i = tid; i < 32 * 64; i += 256) {
    int mrow = i >> 6, mcol = i & 63;
    Wq[mrow * LDW + mcol] = (bf16)qwm[i];
    Wk[mrow * LDW + mcol] = (bf16)kwm[i];
  }
  for (int i = tid; i < 64 * 64; i += 256) {
    int mrow = i >> 6, mcol = i & 63;
    Wv[mrow * LDW + mcol] = (bf16)vwm[i];
    Wo[mrow * LDW + mcol] = (bf16)owm[i];
  }
  for (int i = tid; i < 64 * 256; i += 256) {       // gather group pixels
    int c = i >> 8, pos = i & 255;                  // pos fastest -> w-coalesced
    int qh = pos >> 4, qw = pos & 15;
    XbT[pos * LDP + c] =
        (bf16)r[((n * 64 + c) * 128 + qh * 8 + ph) * 128 + qw * 8 + pw];
  }
  __syncthreads();

  // ---- Q/K/V projections: 32 + 32 + 64 = 128 D-tiles across 8 waves --------
  for (int t = wave; t < 128; t += 8) {
    const bf16* W; const float* bias; bf16* dstT; bf16* dstV; int mt, nt;
    dstT = nullptr; dstV = nullptr;
    if (t < 32)      { W = Wq; bias = qb; dstT = Qt; mt = t >> 4;        nt = t & 15; }
    else if (t < 64) { W = Wk; bias = kb; dstT = Kt; mt = (t - 32) >> 4; nt = t & 15; }
    else             { W = Wv; bias = vb; dstV = Vb; mt = (t - 64) >> 4; nt = t & 15; }
    v8f acc = {};
#pragma unroll
    for (int kc = 0; kc < 2; ++kc) {
      v16bf a = frag_rk(W + (mt * 16) * LDW + kc * 32, LDW);
      v16bf b = frag_rk(XbT + (nt * 16) * LDP + kc * 32, LDP);
      acc = wmma_bf16(a, b, acc);
    }
    const int col = nt * 16 + (lane & 15), hh = lane >> 4;
    if (dstT) {  // transposed store [pos][kc]; 8 consecutive kc -> one b128
      v8bf pk;
#pragma unroll
      for (int i = 0; i < 8; ++i)
        pk[i] = (bf16)(acc[i] + bias[mt * 16 + 8 * hh + i]);
      *(v8bf*)(dstT + col * LDQK + mt * 16 + 8 * hh) = pk;
    } else {     // V kept [v][pos]
#pragma unroll
      for (int i = 0; i < 8; ++i) {
        int row = mt * 16 + i + 8 * hh;
        dstV[row * LDN + col] = (bf16)(acc[i] + bias[row]);
      }
    }
  }
  __syncthreads();

  // ---- attention: each wave owns 2 row-tiles (32 query positions) ----------
  const float scale = 0.17677669529663687f;  // 1/sqrt(KC)
  for (int rt = wave * 2; rt < wave * 2 + 2; ++rt) {
    const int np0 = rt * 16;
    v16bf aq = frag_rk(Qt + np0 * LDQK, LDQK);  // A(np,kc), single K-chunk
    float rmax[8], rsum[8];
#pragma unroll
    for (int i = 0; i < 8; ++i) { rmax[i] = -3.0e38f; rsum[i] = 0.0f; }
    for (int mt = 0; mt < 16; ++mt) {   // pass 1: row max
      v16bf bk = frag_rk(Kt + (mt * 16) * LDQK, LDQK);
      v8f s = wmma_bf16(aq, bk, v8f{});
#pragma unroll
      for (int i = 0; i < 8; ++i) rmax[i] = fmaxf(rmax[i], grp16_max(s[i]));
    }
    for (int mt = 0; mt < 16; ++mt) {   // pass 2: exp (unnormalized) + row sum
      v16bf bk = frag_rk(Kt + (mt * 16) * LDQK, LDQK);
      v8f s = wmma_bf16(aq, bk, v8f{});
      int hh = lane >> 4;
#pragma unroll
      for (int i = 0; i < 8; ++i) {
        float e = __expf((s[i] - rmax[i]) * scale);
        Ab[(np0 + i + 8 * hh) * LDN + mt * 16 + (lane & 15)] = (bf16)e;
        rsum[i] += grp16_sum(e);
      }
    }
    // ctx^T[np][v] = attn x V^T ; normalize by rsum at the end
    for (int vt = 0; vt < 4; ++vt) {
      v8f acc = {};
#pragma unroll
      for (int kc = 0; kc < 8; ++kc) {
        v16bf a = frag_rk(Ab + np0 * LDN + kc * 32, LDN);
        v16bf b = frag_rk(Vb + (vt * 16) * LDN + kc * 32, LDN);
        acc = wmma_bf16(a, b, acc);
      }
      int hh = lane >> 4;
#pragma unroll
      for (int i = 0; i < 8; ++i) {
        int row = np0 + i + 8 * hh;       // position
        int v = vt * 16 + (lane & 15);    // value channel
        Ct[row * LDP + v] = (bf16)(acc[i] / rsum[i]);
      }
    }
  }
  __syncthreads();

  // ---- output projection + in-place scatter back -------------------------
  for (int t = wave; t < 64; t += 8) {
    int mt = t >> 4, nt = t & 15;
    v8f acc = {};
#pragma unroll
    for (int kc = 0; kc < 2; ++kc) {
      v16bf a = frag_rk(Wo + (mt * 16) * LDW + kc * 32, LDW);
      v16bf b = frag_rk(Ct + (nt * 16) * LDP + kc * 32, LDP);
      acc = wmma_bf16(a, b, acc);
    }
    int pos = nt * 16 + (lane & 15);
    int qh = pos >> 4, qw = pos & 15, hh = lane >> 4;
#pragma unroll
    for (int i = 0; i < 8; ++i) {
      int o = mt * 16 + i + 8 * hh;
      r[((n * 64 + o) * 128 + qh * 8 + ph) * 128 + qw * 8 + pw] = acc[i] + ob[o];
    }
  }
}

// ---------------------------------------------------------------------------
// Kernel C: short-range SA over 8x8 windows + residual (out = x + SA(r)).
// Grid: 4096 blocks (one per window); 128 threads = 4 waves.
// ---------------------------------------------------------------------------
#define SLDN 72   // 64 positions + 8 pad

__global__ __launch_bounds__(128)
void isa_short_kernel(const float* __restrict__ r, const float* __restrict__ x,
                      const float* __restrict__ qwm, const float* __restrict__ qb,
                      const float* __restrict__ kwm, const float* __restrict__ kb,
                      const float* __restrict__ vwm, const float* __restrict__ vb,
                      const float* __restrict__ owm, const float* __restrict__ ob,
                      float* __restrict__ out) {
  extern __shared__ char smem[];
  bf16* Wq  = (bf16*)smem;           // 32 x LDW
  bf16* Wk  = Wq + 32 * LDW;
  bf16* Wv  = Wk + 32 * LDW;         // 64 x LDW
  bf16* Wo  = Wv + 64 * LDW;
  bf16* XbT = Wo + 64 * LDW;         // 64 x LDP   [pos][c] (aliased by Ct)
  bf16* Qt  = XbT + 64 * LDP;        // 64 x LDQK  [pos][kc]
  bf16* Kt  = Qt + 64 * LDQK;        // 64 x LDQK
  bf16* Vb  = Kt + 64 * LDQK;        // 64 x SLDN  [v][pos]
  bf16* Ab  = Vb + 64 * SLDN;        // 64 x SLDN  [np][m]
  bf16* Ct  = XbT;

  const int g = blockIdx.x;
  const int n = g >> 8, qh = (g >> 4) & 15, qw = g & 15;
  const int tid = threadIdx.x, lane = tid & 31, wave = tid >> 5;

  for (int i = tid; i < 32 * 64; i += 128) {
    int mrow = i >> 6, mcol = i & 63;
    Wq[mrow * LDW + mcol] = (bf16)qwm[i];
    Wk[mrow * LDW + mcol] = (bf16)kwm[i];
  }
  for (int i = tid; i < 64 * 64; i += 128) {
    int mrow = i >> 6, mcol = i & 63;
    Wv[mrow * LDW + mcol] = (bf16)vwm[i];
    Wo[mrow * LDW + mcol] = (bf16)owm[i];
  }
  for (int i = tid; i < 64 * 64; i += 128) {
    int c = i >> 6, pos = i & 63;
    XbT[pos * LDP + c] =
        (bf16)r[((n * 64 + c) * 128 + qh * 8 + (pos >> 3)) * 128 + qw * 8 + (pos & 7)];
  }
  __syncthreads();

  // ---- projections: 8 + 8 + 16 = 32 D-tiles across 4 waves ----------------
  for (int t = wave; t < 32; t += 4) {
    const bf16* W; const float* bias; bf16* dstT; bf16* dstV; int mt, nt;
    dstT = nullptr; dstV = nullptr;
    if (t < 8)       { W = Wq; bias = qb; dstT = Qt; mt = t >> 2;        nt = t & 3; }
    else if (t < 16) { W = Wk; bias = kb; dstT = Kt; mt = (t - 8) >> 2;  nt = t & 3; }
    else             { W = Wv; bias = vb; dstV = Vb; mt = (t - 16) >> 2; nt = t & 3; }
    v8f acc = {};
#pragma unroll
    for (int kc = 0; kc < 2; ++kc) {
      v16bf a = frag_rk(W + (mt * 16) * LDW + kc * 32, LDW);
      v16bf b = frag_rk(XbT + (nt * 16) * LDP + kc * 32, LDP);
      acc = wmma_bf16(a, b, acc);
    }
    const int col = nt * 16 + (lane & 15), hh = lane >> 4;
    if (dstT) {
      v8bf pk;
#pragma unroll
      for (int i = 0; i < 8; ++i)
        pk[i] = (bf16)(acc[i] + bias[mt * 16 + 8 * hh + i]);
      *(v8bf*)(dstT + col * LDQK + mt * 16 + 8 * hh) = pk;
    } else {
#pragma unroll
      for (int i = 0; i < 8; ++i) {
        int row = mt * 16 + i + 8 * hh;
        dstV[row * SLDN + col] = (bf16)(acc[i] + bias[row]);
      }
    }
  }
  __syncthreads();

  // ---- attention: each wave owns 1 row-tile (16 positions) ----------------
  const float scale = 0.17677669529663687f;
  {
    const int np0 = wave * 16;
    v16bf aq = frag_rk(Qt + np0 * LDQK, LDQK);
    float rmax[8], rsum[8];
#pragma unroll
    for (int i = 0; i < 8; ++i) { rmax[i] = -3.0e38f; rsum[i] = 0.0f; }
    for (int mt = 0; mt < 4; ++mt) {
      v16bf bk = frag_rk(Kt + (mt * 16) * LDQK, LDQK);
      v8f s = wmma_bf16(aq, bk, v8f{});
#pragma unroll
      for (int i = 0; i < 8; ++i) rmax[i] = fmaxf(rmax[i], grp16_max(s[i]));
    }
    for (int mt = 0; mt < 4; ++mt) {
      v16bf bk = frag_rk(Kt + (mt * 16) * LDQK, LDQK);
      v8f s = wmma_bf16(aq, bk, v8f{});
      int hh = lane >> 4;
#pragma unroll
      for (int i = 0; i < 8; ++i) {
        float e = __expf((s[i] - rmax[i]) * scale);
        Ab[(np0 + i + 8 * hh) * SLDN + mt * 16 + (lane & 15)] = (bf16)e;
        rsum[i] += grp16_sum(e);
      }
    }
    for (int vt = 0; vt < 4; ++vt) {
      v8f acc = {};
#pragma unroll
      for (int kc = 0; kc < 2; ++kc) {
        v16bf a = frag_rk(Ab + np0 * SLDN + kc * 32, SLDN);
        v16bf b = frag_rk(Vb + (vt * 16) * SLDN + kc * 32, SLDN);
        acc = wmma_bf16(a, b, acc);
      }
      int hh = lane >> 4;
#pragma unroll
      for (int i = 0; i < 8; ++i) {
        int row = np0 + i + 8 * hh;
        int v = vt * 16 + (lane & 15);
        Ct[row * LDP + v] = (bf16)(acc[i] / rsum[i]);
      }
    }
  }
  __syncthreads();

  // ---- output projection + residual ---------------------------------------
  for (int t = wave; t < 16; t += 4) {
    int mt = t >> 2, nt = t & 3;
    v8f acc = {};
#pragma unroll
    for (int kc = 0; kc < 2; ++kc) {
      v16bf a = frag_rk(Wo + (mt * 16) * LDW + kc * 32, LDW);
      v16bf b = frag_rk(Ct + (nt * 16) * LDP + kc * 32, LDP);
      acc = wmma_bf16(a, b, acc);
    }
    int pos = nt * 16 + (lane & 15), hh = lane >> 4;
    int H = qh * 8 + (pos >> 3), W = qw * 8 + (pos & 7);
#pragma unroll
    for (int i = 0; i < 8; ++i) {
      int o = mt * 16 + i + 8 * hh;
      size_t idx = ((size_t)(n * 64 + o) * 128 + H) * 128 + W;
      out[idx] = x[idx] + acc[i] + ob[o];
    }
  }
}

// ---------------------------------------------------------------------------
extern "C" void kernel_launch(void* const* d_in, const int* in_sizes, int n_in,
                              void* d_out, int out_size, void* d_ws, size_t ws_size,
                              hipStream_t stream) {
  const float* x       = (const float*)d_in[0];
  const float* conv1_w = (const float*)d_in[1];
  const float* conv1_b = (const float*)d_in[2];
  const float* prelu_a = (const float*)d_in[3];
  const float* lqw = (const float*)d_in[4];  const float* lqb = (const float*)d_in[5];
  const float* lkw = (const float*)d_in[6];  const float* lkb = (const float*)d_in[7];
  const float* lvw = (const float*)d_in[8];  const float* lvb = (const float*)d_in[9];
  const float* lowm = (const float*)d_in[10]; const float* lob = (const float*)d_in[11];
  const float* sqw = (const float*)d_in[12]; const float* sqb = (const float*)d_in[13];
  const float* skw = (const float*)d_in[14]; const float* skb = (const float*)d_in[15];
  const float* svw = (const float*)d_in[16]; const float* svb = (const float*)d_in[17];
  const float* sow = (const float*)d_in[18]; const float* sob = (const float*)d_in[19];
  float* out = (float*)d_out;
  float* r = (float*)d_ws;  // 16*64*128*128 f32 = 64 MiB intermediate

  const size_t ldsA = (size_t)(3 * 130 * CONV_LDX + 64 * CONV_LDW) * sizeof(bf16);
  const size_t ldsB = (size_t)(2 * 32 * LDW + 2 * 64 * LDW + 256 * LDP +
                               2 * 256 * LDQK + 64 * LDN + 256 * LDN) * sizeof(bf16);
  const size_t ldsC = (size_t)(2 * 32 * LDW + 2 * 64 * LDW + 64 * LDP +
                               2 * 64 * LDQK + 64 * SLDN + 64 * SLDN) * sizeof(bf16);

  (void)hipFuncSetAttribute((const void*)conv_prelu_kernel,
                            hipFuncAttributeMaxDynamicSharedMemorySize, (int)ldsA);
  (void)hipFuncSetAttribute((const void*)isa_long_kernel,
                            hipFuncAttributeMaxDynamicSharedMemorySize, (int)ldsB);
  (void)hipFuncSetAttribute((const void*)isa_short_kernel,
                            hipFuncAttributeMaxDynamicSharedMemorySize, (int)ldsC);

  conv_prelu_kernel<<<16 * 128, 256, ldsA, stream>>>(x, conv1_w, conv1_b, prelu_a, r);
  isa_long_kernel<<<1024, 256, ldsB, stream>>>(r, lqw, lqb, lkw, lkb, lvw, lvb, lowm, lob);
  isa_short_kernel<<<4096, 128, ldsC, stream>>>(r, x, sqw, sqb, skw, skb, svw, svb, sow, sob, out);
}
